// SO2EdgeUpdate_27444841021660
// MI455X (gfx1250) — matrix operand order repaired and checked by reading, not compiled, not run on hardware
//
#include <hip/hip_runtime.h>
#include <cstdint>
#include <cstddef>

// ---------------------------------------------------------------------------
// SO(2) equivariant edge update for MI455X (gfx1250), bf16 WMMA pipeline.
// ~100 GFLOP of GEMM vs ~300MB HBM traffic -> matrix-core bound; use
// v_wmma_f32_16x16x32_bf16 with f32 accumulation everywhere.
// mp intermediate is chunked (29MB) so it stays resident in the 192MB L2.
// ---------------------------------------------------------------------------

typedef __bf16 bf16_t;
typedef __attribute__((ext_vector_type(16))) __bf16 v16bf;
typedef __attribute__((ext_vector_type(8)))  float  v8f;

#define E_TOTAL   10000
#define N_NODES   1000
#define C3        384
#define NCF       25
#define NRED      19
#define HH        64
#define XEDGE     384
#define RAD_H     128
#define WRAD_N    1920      // (L+1)*C3
#define M0_COL0   512       // first live output column of w_m0
#define M0_NUSED  576       // live columns 512..1087
#define M0_LDW    1920
#define M1_N      512
#define M1_K      1536
#define M2_N      384
#define M2_K      1152
#define OUTC      128
#define MP_STRIDE 7296      // 19*384
#define HL_STRIDE 1216      // 19*64
#define TE        16
#define CHUNK     2000

// wigner row for permuted mp row m: MASK[PERM_M[m]]
__constant__ int d_MASK_PERM[NRED] = {0,2,6,12,20, 3,7,13,21, 1,5,11,19, 8,14,22, 4,10,18};
// wigner row for reduced index m (second contraction)
__constant__ int d_MASK[NRED]      = {0,1,2,3,4,5,6,7,8,10,11,12,13,14,18,19,20,21,22};
// reduced index -> permuted row
__constant__ int d_INV_PERM[NRED]  = {0,9,1,5,16,10,2,6,13,17,11,3,7,14,18,12,4,8,15};
// gating l-1 per reduced index 1..18
__constant__ int d_GATE[NRED-1]    = {0,0,0,1,1,1,1,1,2,2,2,2,2,3,3,3,3,3};
// l per full index n (0..24)
__constant__ int d_LOFN[NCF]       = {0,1,1,1,2,2,2,2,2,3,3,3,3,3,3,3,4,4,4,4,4,4,4,4,4};

__device__ __forceinline__ v8f wmma_bf16(v16bf a, v16bf b, v8f c) {
  return __builtin_amdgcn_wmma_f32_16x16x32_bf16(false, a, false, b, (short)0, c, false, false);
}

// A element k-pattern (contiguous, matches B loaders): lanes 0-15 hold K 0..15,
// lanes 16-31 hold K 16..31, 2 K per VGPR.
__device__ __forceinline__ int a_k_pat(int j, int t, int half) {
  return 2*j + t + 16*half;
}

// Load A tile from per-lane row pointer (row-major bf16, row chosen by caller via lane&15)
__device__ __forceinline__ v16bf load_a_row(const bf16_t* rowp, int k0) {
  const int half = (threadIdx.x & 31) >> 4;
  v16bf a;
#pragma unroll
  for (int j = 0; j < 8; ++j)
#pragma unroll
    for (int t = 0; t < 2; ++t)
      a[2*j + t] = rowp[k0 + a_k_pat(j, t, half)];
  return a;
}

// Load B tile from weight stored [n][k] row-major (i.e. B[k][n] = W[n][k])
__device__ __forceinline__ v16bf load_bT(const bf16_t* W, int ldw, int n0, int k0) {
  const int lane = threadIdx.x & 31;
  const bf16_t* p = W + (size_t)(n0 + (lane & 15)) * ldw + k0 + ((lane >> 4) * 16);
  v16bf b;
#pragma unroll
  for (int u = 0; u < 16; ++u) b[u] = p[u];
  return b;
}

// Load B tile transposed from f32 weight [n][k]
__device__ __forceinline__ v16bf load_bT_f32(const float* W, int ldw, int n0, int k0) {
  const int lane = threadIdx.x & 31;
  const float* p = W + (size_t)(n0 + (lane & 15)) * ldw + k0 + ((lane >> 4) * 16);
  v16bf b;
#pragma unroll
  for (int u = 0; u < 16; ++u) b[u] = (bf16_t)p[u];
  return b;
}

// Load B tile from row-major B[k][n] (LDS), lanes 0-15: K 0..15, lanes 16-31: K 16..31
__device__ __forceinline__ v16bf load_b_km(const bf16_t* B, int ldb, int k0, int n0) {
  const int lane = threadIdx.x & 31;
  const int kb = k0 + ((lane >> 4) * 16);
  const int n  = n0 + (lane & 15);
  v16bf b;
#pragma unroll
  for (int u = 0; u < 16; ++u) b[u] = B[(size_t)(kb + u) * ldb + n];
  return b;
}

// Store C/D 16x16 f32 tile (row = r + 8*half, col = lane&15)
__device__ __forceinline__ void store_c_lds(float* Cbase, int ldc, const v8f& acc) {
  const int lane = threadIdx.x & 31;
  const int half = lane >> 4, col = lane & 15;
#pragma unroll
  for (int r = 0; r < 8; ++r) Cbase[(size_t)(r + 8*half) * ldc + col] = acc[r];
}

__device__ __forceinline__ float sigmoidf(float x) { return 1.f / (1.f + __expf(-x)); }

// LDS byte offset of a generic pointer into a __shared__ object
__device__ __forceinline__ uint32_t lds_offset(const void* p) {
  return (uint32_t)(uintptr_t)(__attribute__((address_space(3))) const void*)p;
}

// async copy of one 16B chunk global -> LDS (tracked by ASYNCcnt)
__device__ __forceinline__ void async_g2l_b128(uint32_t lds_addr, const void* gaddr) {
  asm volatile("global_load_async_to_lds_b128 %0, %1, off"
               :: "v"(lds_addr), "v"(gaddr) : "memory");
}

// ---------------------------------------------------------------------------
// f32 -> bf16 weight conversion
// ---------------------------------------------------------------------------
__global__ void k_cvt_bf16(const float* __restrict__ in, bf16_t* __restrict__ outp, int n) {
  int i = blockIdx.x * 256 + threadIdx.x;
  if (i < n) outp[i] = (bf16_t)in[i];
}

// ---------------------------------------------------------------------------
// Kernel B: radial MLP + message scaling + Wigner rotation (permuted) -> mp
// block = 128 threads (4 waves), 16 edges per block
// ---------------------------------------------------------------------------
__global__ void __launch_bounds__(128)
k_radial_wigner(const float* __restrict__ x,
                const float* __restrict__ edge_fea,
                const float* __restrict__ edge_distance,
                const float* __restrict__ wigner,
                const float* __restrict__ src_emb,
                const float* __restrict__ tgt_emb,
                const bf16_t* __restrict__ w1, const float* __restrict__ b1,
                const float* __restrict__ g1, const float* __restrict__ bb1,
                const bf16_t* __restrict__ w2, const float* __restrict__ b2,
                const float* __restrict__ g2, const float* __restrict__ bb2,
                const bf16_t* __restrict__ w3, const float* __restrict__ b3,
                const int* __restrict__ atomic_numbers,
                const int* __restrict__ edge_index,
                bf16_t* __restrict__ mp_out,
                int e_base)
{
  __shared__ bf16_t s_a0[TE][XEDGE];       // x_edge bf16
  __shared__ float  s_h[TE][RAD_H];        // GEMM accum / LN scratch
  __shared__ bf16_t s_a1[TE][RAD_H];       // layer1 act
  __shared__ bf16_t s_a2[TE][RAD_H];       // layer2 act
  __shared__ bf16_t s_wrad[TE][WRAD_N];    // radial weights (5*384 per edge)
  __shared__ bf16_t s_msg[4][32][128];     // per-wave scaled msg column chunk

  const int tid  = threadIdx.x;
  const int lane = tid & 31;
  const int wv   = tid >> 5;
  const int blk16 = blockIdx.x * TE;
  const int e0    = e_base + blk16;

  // zero K-pad rows 25..31 of the per-wave message buffer (stay zero forever)
  for (int idx = lane; idx < 7*128; idx += 32)
    s_msg[wv][25 + (idx >> 7)][idx & 127] = (bf16_t)0.f;

  // ---- gather x_edge = [edge_distance | src_emb | tgt_emb] as bf16 ----
  for (int idx = tid; idx < TE*XEDGE; idx += 128) {
    int el = idx / XEDGE, c = idx % XEDGE;
    int e = e0 + el;
    float v;
    if (c < 128)      v = edge_distance[(size_t)e*128 + c];
    else if (c < 256) { int s = edge_index[e];          v = src_emb[(size_t)atomic_numbers[s]*128 + (c-128)]; }
    else              { int t = edge_index[E_TOTAL + e]; v = tgt_emb[(size_t)atomic_numbers[t]*128 + (c-256)]; }
    s_a0[el][c] = (bf16_t)v;
  }
  __syncthreads();

  // ---- layer 1: (16x384) @ w1^T -> 16x128  (A shared across 2 N-tiles) ----
  {
    const bf16_t* rowp = &s_a0[lane & 15][0];
    const int nt0 = wv*2;
    v8f acc0 = {}, acc1 = {};
#pragma unroll
    for (int kk = 0; kk < XEDGE/32; ++kk) {
      v16bf a = load_a_row(rowp, kk*32);
      acc0 = wmma_bf16(a, load_bT(w1, XEDGE, nt0*16,      kk*32), acc0);
      acc1 = wmma_bf16(a, load_bT(w1, XEDGE, (nt0+1)*16,  kk*32), acc1);
    }
    store_c_lds(&s_h[0][0] + nt0*16,     RAD_H, acc0);
    store_c_lds(&s_h[0][0] + (nt0+1)*16, RAD_H, acc1);
  }
  __syncthreads();

  // ---- LayerNorm + SiLU (8 lanes per edge, 16 channels each) ----
  {
    int el = tid >> 3, sub = tid & 7;
    float vals[16]; float sum = 0.f, sq = 0.f;
#pragma unroll
    for (int i = 0; i < 16; ++i) {
      int c = sub*16 + i;
      float v = s_h[el][c] + b1[c];
      vals[i] = v; sum += v; sq += v*v;
    }
#pragma unroll
    for (int off = 1; off < 8; off <<= 1) { sum += __shfl_xor(sum, off, 8); sq += __shfl_xor(sq, off, 8); }
    float mu = sum * (1.f/128.f), var = sq * (1.f/128.f) - mu*mu;
    float inv = rsqrtf(var + 1e-5f);
#pragma unroll
    for (int i = 0; i < 16; ++i) {
      int c = sub*16 + i;
      float nv = (vals[i] - mu) * inv * g1[c] + bb1[c];
      s_a1[el][c] = (bf16_t)(nv * sigmoidf(nv));
    }
  }
  __syncthreads();

  // ---- layer 2: (16x128) @ w2^T -> 16x128 ----
  {
    const bf16_t* rowp = &s_a1[lane & 15][0];
    const int nt0 = wv*2;
    v8f acc0 = {}, acc1 = {};
#pragma unroll
    for (int kk = 0; kk < RAD_H/32; ++kk) {
      v16bf a = load_a_row(rowp, kk*32);
      acc0 = wmma_bf16(a, load_bT(w2, RAD_H, nt0*16,     kk*32), acc0);
      acc1 = wmma_bf16(a, load_bT(w2, RAD_H, (nt0+1)*16, kk*32), acc1);
    }
    store_c_lds(&s_h[0][0] + nt0*16,     RAD_H, acc0);
    store_c_lds(&s_h[0][0] + (nt0+1)*16, RAD_H, acc1);
  }
  __syncthreads();
  {
    int el = tid >> 3, sub = tid & 7;
    float vals[16]; float sum = 0.f, sq = 0.f;
#pragma unroll
    for (int i = 0; i < 16; ++i) {
      int c = sub*16 + i;
      float v = s_h[el][c] + b2[c];
      vals[i] = v; sum += v; sq += v*v;
    }
#pragma unroll
    for (int off = 1; off < 8; off <<= 1) { sum += __shfl_xor(sum, off, 8); sq += __shfl_xor(sq, off, 8); }
    float mu = sum * (1.f/128.f), var = sq * (1.f/128.f) - mu*mu;
    float inv = rsqrtf(var + 1e-5f);
#pragma unroll
    for (int i = 0; i < 16; ++i) {
      int c = sub*16 + i;
      float nv = (vals[i] - mu) * inv * g2[c] + bb2[c];
      s_a2[el][c] = (bf16_t)(nv * sigmoidf(nv));
    }
  }
  __syncthreads();

  // ---- layer 3: (16x128) @ w3^T -> 16x1920 (A hoisted: 4 K-tiles, reused 30x) ----
  {
    const bf16_t* rowp = &s_a2[lane & 15][0];
    v16bf a3[4];
#pragma unroll
    for (int kk = 0; kk < 4; ++kk) a3[kk] = load_a_row(rowp, kk*32);
    for (int nt = wv*30; nt < wv*30 + 30; ++nt) {
      v8f acc = {};
#pragma unroll
      for (int kk = 0; kk < 4; ++kk)
        acc = wmma_bf16(a3[kk], load_bT(w3, RAD_H, nt*16, kk*32), acc);
      const int col = lane & 15, hf = lane >> 4;
#pragma unroll
      for (int r = 0; r < 8; ++r) {
        int c = nt*16 + col;
        s_wrad[r + 8*hf][c] = (bf16_t)(acc[r] + b3[c]);
      }
    }
  }
  __syncthreads();

  // ---- per-edge Wigner rotation: mp(19x384) = Wperm(19x25) @ (msg .* wrad) ----
  for (int el = wv; el < TE; el += 4) {
    const int e  = e0 + el;
    const int sn = edge_index[e];
    const int tn = edge_index[E_TOTAL + e];
    const float* wig_e = wigner + (size_t)e * (NCF*NCF);

    // A tiles (wigner rows in MASK∘PERM order, zero padded)
    v16bf aw[2];
#pragma unroll
    for (int mt = 0; mt < 2; ++mt) {
      const int half = lane >> 4;
      const int mrow = mt*16 + (lane & 15);
      v16bf a;
#pragma unroll
      for (int j = 0; j < 8; ++j)
#pragma unroll
        for (int t = 0; t < 2; ++t) {
          int k = a_k_pat(j, t, half);
          float v = (mrow < NRED && k < NCF) ? wig_e[d_MASK_PERM[mrow]*NCF + k] : 0.f;
          a[2*j + t] = (bf16_t)v;
        }
      aw[mt] = a;
    }

    for (int cb = 0; cb < 3; ++cb) {
      const int c0 = cb * 128;
      // build scaled message chunk (25 rows x 128 cols) in LDS
      for (int idx = lane; idx < NCF*128; idx += 32) {
        int k = idx >> 7, cc = idx & 127, gc = c0 + cc;
        float f;
        if (gc < 128)      f = x[((size_t)sn*NCF + k)*128 + gc];
        else if (gc < 256) f = x[((size_t)tn*NCF + k)*128 + (gc - 128)];
        else               f = edge_fea[((size_t)e*NCF + k)*128 + (gc - 256)];
        float wr = (float)s_wrad[el][d_LOFN[k]*C3 + gc];
        s_msg[wv][k][cc] = (bf16_t)(f * wr);
      }
      asm volatile("s_wait_dscnt 0" ::: "memory");

#pragma unroll
      for (int nt = 0; nt < 8; ++nt) {
        v16bf b = load_b_km(&s_msg[wv][0][0], 128, 0, nt*16);
        v8f acc0 = {}; acc0 = wmma_bf16(aw[0], b, acc0);
        v8f acc1 = {}; acc1 = wmma_bf16(aw[1], b, acc1);
        const int col = lane & 15, hf = lane >> 4;
        bf16_t* mpe = mp_out + (size_t)(blk16 + el) * MP_STRIDE;
#pragma unroll
        for (int r = 0; r < 8; ++r) {
          int m = r + 8*hf;
          mpe[(size_t)m*C3 + c0 + nt*16 + col] = (bf16_t)acc0[r];
          int m2 = m + 16;
          if (m2 < NRED) mpe[(size_t)m2*C3 + c0 + nt*16 + col] = (bf16_t)acc1[r];
        }
      }
    }
  }
}

// ---------------------------------------------------------------------------
// Kernel C: SO(2) linear layers (w_m0 live cols / w_m1 / w_m2) + gating -> hl
// block = 256 threads (8 waves), 16 edges per block. N-tiles paired so each
// A tile feeds two v_wmma.
// ---------------------------------------------------------------------------
__global__ void __launch_bounds__(256)
k_so2(const bf16_t* __restrict__ mp, const bf16_t* __restrict__ wm0,
      const float* __restrict__ bm0, const bf16_t* __restrict__ wm1,
      const bf16_t* __restrict__ wm2, bf16_t* __restrict__ hl_out, int e_base)
{
  __shared__ float s_x0[TE][M0_NUSED];    // x0 cols 512..1087 (gating + m=0 out)
  __shared__ float s_y1[2*TE][M1_N];      // ym1 rows (e,j)
  __shared__ float s_y2[2*TE][M2_N];      // ym2 rows (e,j)

  const int tid  = threadIdx.x;
  const int lane = tid & 31;
  const int wv   = tid >> 5;
  const int blk16 = blockIdx.x * TE;
  (void)e_base;

  // jobs: 18 (gemm0, N-pairs) + 32 (gemm1: 2 Mtiles x 16 N-pairs) + 24 (gemm2: 2 x 12)
  for (int job = wv; job < 74; job += 8) {
    const int col = lane & 15, hf = lane >> 4;
    if (job < 18) {
      const int nt0 = job * 2;
      const bf16_t* rowp = mp + (size_t)(blk16 + (lane & 15)) * MP_STRIDE;  // rows 0..4 flat = 1920
      const bf16_t* wp = wm0 + (size_t)M0_COL0 * M0_LDW;
      __builtin_prefetch(wp + (size_t)nt0*16*M0_LDW, 0, 1);
      v8f acc0 = {}, acc1 = {};
      for (int kk = 0; kk < M0_LDW/32; ++kk) {
        v16bf a = load_a_row(rowp, kk*32);
        acc0 = wmma_bf16(a, load_bT(wp, M0_LDW, nt0*16,     kk*32), acc0);
        acc1 = wmma_bf16(a, load_bT(wp, M0_LDW, (nt0+1)*16, kk*32), acc1);
      }
#pragma unroll
      for (int r = 0; r < 8; ++r) {
        int n0 = nt0*16 + col;
        s_x0[r + 8*hf][n0]      = acc0[r] + bm0[M0_COL0 + n0];
        s_x0[r + 8*hf][n0 + 16] = acc1[r] + bm0[M0_COL0 + n0 + 16];
      }
    } else if (job < 50) {
      const int j  = job - 18;
      const int mt = j & 1, nt0 = (j >> 1) * 2;
      const int R  = mt*16 + (lane & 15);                   // R = 2*el + jrow
      const bf16_t* rowp = mp + (size_t)(blk16 + (R >> 1)) * MP_STRIDE + (5 + 4*(R & 1)) * C3;
      __builtin_prefetch(wm1 + (size_t)nt0*16*M1_K, 0, 1);
      v8f acc0 = {}, acc1 = {};
      for (int kk = 0; kk < M1_K/32; ++kk) {
        v16bf a = load_a_row(rowp, kk*32);
        acc0 = wmma_bf16(a, load_bT(wm1, M1_K, nt0*16,     kk*32), acc0);
        acc1 = wmma_bf16(a, load_bT(wm1, M1_K, (nt0+1)*16, kk*32), acc1);
      }
#pragma unroll
      for (int r = 0; r < 8; ++r) {
        s_y1[mt*16 + r + 8*hf][nt0*16 + col]      = acc0[r];
        s_y1[mt*16 + r + 8*hf][nt0*16 + col + 16] = acc1[r];
      }
    } else {
      const int j  = job - 50;
      const int mt = j & 1, nt0 = (j >> 1) * 2;
      const int R  = mt*16 + (lane & 15);
      const bf16_t* rowp = mp + (size_t)(blk16 + (R >> 1)) * MP_STRIDE + (13 + 3*(R & 1)) * C3;
      __builtin_prefetch(wm2 + (size_t)nt0*16*M2_K, 0, 1);
      v8f acc0 = {}, acc1 = {};
      for (int kk = 0; kk < M2_K/32; ++kk) {
        v16bf a = load_a_row(rowp, kk*32);
        acc0 = wmma_bf16(a, load_bT(wm2, M2_K, nt0*16,     kk*32), acc0);
        acc1 = wmma_bf16(a, load_bT(wm2, M2_K, (nt0+1)*16, kk*32), acc1);
      }
#pragma unroll
      for (int r = 0; r < 8; ++r) {
        s_y2[mt*16 + r + 8*hf][nt0*16 + col]      = acc0[r];
        s_y2[mt*16 + r + 8*hf][nt0*16 + col + 16] = acc1[r];
      }
    }
  }
  __syncthreads();

  // combine complex pairs + inverse permutation + SiLU / sigmoid gating
  for (int idx = tid; idx < TE*NRED*HH; idx += 256) {
    int el = idx / (NRED*HH);
    int r  = (idx / HH) % NRED;
    int c  = idx % HH;
    int pr = d_INV_PERM[r];
    float v;
    if (pr < 5)       v = s_x0[el][256 + pr*64 + c];
    else if (pr < 9)  v = s_y1[2*el  ][(pr-5)*64 + c]  - s_y1[2*el+1][256 + (pr-5)*64 + c];
    else if (pr < 13) v = s_y1[2*el+1][(pr-9)*64 + c]  + s_y1[2*el  ][256 + (pr-9)*64 + c];
    else if (pr < 16) v = s_y2[2*el  ][(pr-13)*64 + c] - s_y2[2*el+1][192 + (pr-13)*64 + c];
    else              v = s_y2[2*el+1][(pr-16)*64 + c] + s_y2[2*el  ][192 + (pr-16)*64 + c];
    float outv;
    if (r == 0) outv = v * sigmoidf(v);                       // SiLU on l=0,m=0
    else        outv = v * sigmoidf(s_x0[el][d_GATE[r-1]*64 + c]);
    hl_out[(size_t)(blk16 + el) * HL_STRIDE + r*64 + c] = (bf16_t)outv;
  }
}

// ---------------------------------------------------------------------------
// Kernel D: inverse Wigner (full = W^T @ hl) + per-l projection -> out
// block = 256 threads (8 waves), 16 edges per block.
// hl is staged to LDS via async global->LDS b128 copies (ASYNCcnt path).
// ---------------------------------------------------------------------------
__global__ void __launch_bounds__(256)
k_project(const float* __restrict__ wigner, const bf16_t* __restrict__ hl,
          const float* __restrict__ proj_w, const float* __restrict__ proj_b,
          float* __restrict__ out, int e_base)
{
  __shared__ bf16_t s_hl[8][32][HH];        // per-wave staged hl, zero padded
  __shared__ bf16_t s_full[TE][NCF][HH];    // rotated features, all 16 edges

  const int tid  = threadIdx.x;
  const int lane = tid & 31;
  const int wv   = tid >> 5;
  const int blk16 = blockIdx.x * TE;

  // zero K-pad rows 19..31 once (never overwritten)
  for (int idx = NRED*HH + lane; idx < 32*HH; idx += 32)
    s_hl[wv][idx >> 6][idx & 63] = (bf16_t)0.f;

  // ---- phase 1: full(25x64) = Wred^T(25x19) @ hl(19x64), per edge ----
  for (int el = wv; el < TE; el += 8) {
    const int e_loc = blk16 + el;
    const int e     = e_base + e_loc;

    // async-copy 19*64 bf16 = 2432B = 152 x 16B chunks into LDS
    {
      const uint32_t lbase = lds_offset(&s_hl[wv][0][0]);
      const char* gbase = (const char*)(hl + (size_t)e_loc * HL_STRIDE);
      for (int chunk = lane; chunk < 152; chunk += 32)
        async_g2l_b128(lbase + chunk*16, gbase + chunk*16);
      asm volatile("s_wait_asynccnt 0" ::: "memory");
      asm volatile("s_wait_dscnt 0" ::: "memory");
    }

    const float* wig_e = wigner + (size_t)e * (NCF*NCF);
#pragma unroll
    for (int mt = 0; mt < 2; ++mt) {
      // A[n][m] = wigner[e][MASK[m]][n], zero padded
      const int half = lane >> 4;
      const int nrow = mt*16 + (lane & 15);
      v16bf a;
#pragma unroll
      for (int j = 0; j < 8; ++j)
#pragma unroll
        for (int t = 0; t < 2; ++t) {
          int m = a_k_pat(j, t, half);
          float v = (nrow < NCF && m < NRED) ? wig_e[d_MASK[m]*NCF + nrow] : 0.f;
          a[2*j + t] = (bf16_t)v;
        }
#pragma unroll
      for (int ct = 0; ct < 4; ++ct) {
        v16bf b = load_b_km(&s_hl[wv][0][0], HH, 0, ct*16);
        v8f acc = {}; acc = wmma_bf16(a, b, acc);
        const int col = lane & 15, hf = lane >> 4;
#pragma unroll
        for (int r = 0; r < 8; ++r) {
          int n = mt*16 + r + 8*hf;
          if (n < NCF) s_full[el][n][ct*16 + col] = (bf16_t)acc[r];
        }
      }
    }
  }
  __syncthreads();

  // ---- phase 2: out[n] = full[n] @ proj_w[l(n)]^T, batched per l group ----
  // jobs: 25 global Mtiles x 4 N-pairs; A tile shared by two v_wmma
  for (int job = wv; job < 100; job += 8) {
    const int mtg = job >> 2;   // 0..24 global Mtile
    const int nt0 = (job & 3) * 2;
    int l;
    if (mtg >= 16) l = 4; else if (mtg >= 9) l = 3; else if (mtg >= 4) l = 2;
    else if (mtg >= 1) l = 1; else l = 0;
    const int cum[5] = {0, 1, 4, 9, 16};
    const int mt_in = mtg - cum[l];
    const int span  = 2*l + 1;

    const int R  = mt_in*16 + (lane & 15);
    const int el = R / span, nl = R % span;
    const bf16_t* rowp = &s_full[el][cum[l] + nl][0];
    const float* pw = proj_w + (size_t)l*OUTC*HH;
    v8f acc0 = {}, acc1 = {};
#pragma unroll
    for (int kk = 0; kk < 2; ++kk) {
      v16bf a = load_a_row(rowp, kk*32);
      acc0 = wmma_bf16(a, load_bT_f32(pw, HH, nt0*16,     kk*32), acc0);
      acc1 = wmma_bf16(a, load_bT_f32(pw, HH, (nt0+1)*16, kk*32), acc1);
    }

    const int col = lane & 15, hf = lane >> 4;
#pragma unroll
    for (int r = 0; r < 8; ++r) {
      const int Rr  = mt_in*16 + r + 8*hf;
      const int elr = Rr / span, nlr = Rr % span;
      const int n   = cum[l] + nlr;
      const int o0  = nt0*16 + col;
      float* op = out + ((size_t)(e_base + blk16 + elr)*NCF + n)*OUTC;
      op[o0]      = acc0[r] + (n == 0 ? proj_b[o0]      : 0.f);
      op[o0 + 16] = acc1[r] + (n == 0 ? proj_b[o0 + 16] : 0.f);
    }
  }
}

// ---------------------------------------------------------------------------
// host launcher
// ---------------------------------------------------------------------------
extern "C" void kernel_launch(void* const* d_in, const int* in_sizes, int n_in,
                              void* d_out, int out_size, void* d_ws, size_t ws_size,
                              hipStream_t stream) {
  (void)in_sizes; (void)n_in; (void)out_size; (void)ws_size;
  const float* x             = (const float*)d_in[0];
  const float* edge_fea      = (const float*)d_in[1];
  const float* edge_distance = (const float*)d_in[2];
  const float* wigner        = (const float*)d_in[3];
  const float* src_emb       = (const float*)d_in[4];
  const float* tgt_emb       = (const float*)d_in[5];
  const float* rad_w1        = (const float*)d_in[6];
  const float* rad_b1        = (const float*)d_in[7];
  const float* rad_g1        = (const float*)d_in[8];
  const float* rad_bb1       = (const float*)d_in[9];
  const float* rad_w2        = (const float*)d_in[10];
  const float* rad_b2        = (const float*)d_in[11];
  const float* rad_g2        = (const float*)d_in[12];
  const float* rad_bb2       = (const float*)d_in[13];
  const float* rad_w3        = (const float*)d_in[14];
  const float* rad_b3        = (const float*)d_in[15];
  const float* w_m0          = (const float*)d_in[16];
  const float* b_m0          = (const float*)d_in[17];
  const float* w_m1          = (const float*)d_in[18];
  const float* w_m2          = (const float*)d_in[19];
  const float* proj_w        = (const float*)d_in[20];
  const float* proj_b        = (const float*)d_in[21];
  const int*   atomic_numbers= (const int*)d_in[22];
  const int*   edge_index    = (const int*)d_in[23];
  float* out = (float*)d_out;

  char* ws = (char*)d_ws;
  size_t off = 0;
  auto alloc = [&](size_t bytes) -> void* {
    off = (off + 255) & ~(size_t)255;
    void* p = ws + off;
    off += bytes;
    return p;
  };

  const int n_w1  = RAD_H * XEDGE;        // 49152
  const int n_w2  = RAD_H * RAD_H;        // 16384
  const int n_w3  = WRAD_N * RAD_H;       // 245760
  const int n_wm0 = 1088 * M0_LDW;        // 2088960
  const int n_wm1 = M1_N * M1_K;          // 786432
  const int n_wm2 = M2_N * M2_K;          // 442368

  bf16_t* bw1  = (bf16_t*)alloc((size_t)n_w1  * sizeof(bf16_t));
  bf16_t* bw2  = (bf16_t*)alloc((size_t)n_w2  * sizeof(bf16_t));
  bf16_t* bw3  = (bf16_t*)alloc((size_t)n_w3  * sizeof(bf16_t));
  bf16_t* bwm0 = (bf16_t*)alloc((size_t)n_wm0 * sizeof(bf16_t));
  bf16_t* bwm1 = (bf16_t*)alloc((size_t)n_wm1 * sizeof(bf16_t));
  bf16_t* bwm2 = (bf16_t*)alloc((size_t)n_wm2 * sizeof(bf16_t));
  bf16_t* mp   = (bf16_t*)alloc((size_t)CHUNK * MP_STRIDE * sizeof(bf16_t));  // ~29 MB (L2 resident)
  bf16_t* hlb  = (bf16_t*)alloc((size_t)CHUNK * HL_STRIDE * sizeof(bf16_t));  // ~4.9 MB

  auto cvt = [&](const float* src, bf16_t* dst, int n) {
    k_cvt_bf16<<<(n + 255) / 256, 256, 0, stream>>>(src, dst, n);
  };
  cvt(rad_w1, bw1, n_w1);
  cvt(rad_w2, bw2, n_w2);
  cvt(rad_w3, bw3, n_w3);
  cvt(w_m0,  bwm0, n_wm0);
  cvt(w_m1,  bwm1, n_wm1);
  cvt(w_m2,  bwm2, n_wm2);

  for (int cb = 0; cb < E_TOTAL; cb += CHUNK) {
    k_radial_wigner<<<CHUNK / TE, 128, 0, stream>>>(
        x, edge_fea, edge_distance, wigner, src_emb, tgt_emb,
        bw1, rad_b1, rad_g1, rad_bb1, bw2, rad_b2, rad_g2, rad_bb2,
        bw3, rad_b3, atomic_numbers, edge_index, mp, cb);
    k_so2<<<CHUNK / TE, 256, 0, stream>>>(mp, bwm0, b_m0, bwm1, bwm2, hlb, cb);
    k_project<<<CHUNK / TE, 256, 0, stream>>>(wigner, hlb, proj_w, proj_b, out, cb);
  }
}
